// Model_69561290326291
// MI455X (gfx1250) — matrix operand order repaired and checked by reading, not compiled, not run on hardware
//
#include <hip/hip_runtime.h>
#include <hip/hip_bf16.h>

typedef __attribute__((ext_vector_type(16))) _Float16 v16h;
typedef __attribute__((ext_vector_type(8)))  _Float16 v8h;
typedef __attribute__((ext_vector_type(8)))  float    v8f;

union V8F  { v8f  v; float    e[8];  };
union V16H { v16h v; _Float16 e[16]; };

__device__ __forceinline__ float sigmoidf_(float x) { return 1.0f / (1.0f + __expf(-x)); }

// A-fragment for v_wmma_f32_16x16x32_f16 from row-major f16:
// lane holds M=l15, K = kb..kb+15 contiguous, kb = khalf*16.
__device__ __forceinline__ v16h load_a_f16(const _Float16* __restrict__ row,
                                           int k0, int khalf)
{
    v8h alo = *(const v8h*)(row + k0 + khalf * 16);
    v8h ahi = *(const v8h*)(row + k0 + khalf * 16 + 8);
    return __builtin_shufflevector(alo, ahi,
                                   0,1,2,3,4,5,6,7,8,9,10,11,12,13,14,15);
}

// ---------------------------------------------------------------------------
// Fused LSTM-step GEMM:  C[M,N] = bias + A1[M,K1]*B1[N,K1]^T + A2[M,K2]*B2[N,K2]^T
// (A2/B2 pass skipped when K2==0, i.e. t==0 where h==0).
// One wave computes a 16x32 output strip (two 16x16 tiles) so each A fragment
// feeds two v_wmma ops.  A_MODE (compile-time):
//   0 = f16 A1, row m at abase1 + m*lda1
//   1 = f16 A1, row m at Aidx[m]*aidxmul            (reversed-seq gather)
//   2 = f32 A1 row = Agather + Aidx[m*ids_stride+ids_off]*lda1  (embedding gather)
// ---------------------------------------------------------------------------
template<int A_MODE>
__global__ void __launch_bounds__(128)
fused_wmma_gemm(const _Float16* __restrict__ A1,
                const float*    __restrict__ Agather,
                const int*      __restrict__ Aidx,
                long long abase1, int lda1, int aidxmul, int ids_stride, int ids_off,
                const _Float16* __restrict__ B1, int ldb1, int K1,
                const _Float16* __restrict__ A2, int lda2,
                const _Float16* __restrict__ B2, int ldb2, int K2,
                float* __restrict__ C, int ldc,
                const float* __restrict__ bias,
                int Mtiles)
{
    const int lane  = threadIdx.x;                       // 0..31 (wave32)
    const int mtile = blockIdx.y * blockDim.y + threadIdx.y;
    if (mtile >= Mtiles) return;                         // uniform across wave

    const int l15   = lane & 15;
    const int khalf = lane >> 4;
    const int m_global = mtile * 16 + l15;
    const int n0 = blockIdx.x * 32 + l15;                // first tile column
    const int n1 = n0 + 16;                              // second tile column

    // accumulators (bias-initialized)
    V8F acc0, acc1;
    {
        float b0 = bias[n0], b1 = bias[n1];
        #pragma unroll
        for (int r = 0; r < 8; ++r) { acc0.e[r] = b0; acc1.e[r] = b1; }
    }

    // ---- pass 1: input projection ----
    const _Float16* b1row0 = B1 + (long long)n0 * ldb1;
    const _Float16* b1row1 = B1 + (long long)n1 * ldb1;

    if (A_MODE == 2) {
        int id = Aidx[(long long)m_global * ids_stride + ids_off];
        const float* ArowF = Agather + (long long)id * lda1;
        for (int k0 = 0; k0 < K1; k0 += 32) {
            V16H at;
            const float* p0 = ArowF + k0 + khalf * 16;
            const float* p1 = ArowF + k0 + khalf * 16 + 8;
            #pragma unroll
            for (int j = 0; j < 8; ++j) {
                at.e[j]     = (_Float16)p0[j];
                at.e[8 + j] = (_Float16)p1[j];
            }
            v16h b0 = *(const v16h*)(b1row0 + k0 + khalf * 16);
            v16h b1 = *(const v16h*)(b1row1 + k0 + khalf * 16);
            acc0.v = __builtin_amdgcn_wmma_f32_16x16x32_f16(
                         false, at.v, false, b0, (short)0, acc0.v, false, false);
            acc1.v = __builtin_amdgcn_wmma_f32_16x16x32_f16(
                         false, at.v, false, b1, (short)0, acc1.v, false, false);
        }
    } else {
        const _Float16* Arow =
            (A_MODE == 0) ? (A1 + abase1 + (long long)m_global * lda1)
                          : (A1 + (long long)Aidx[m_global] * aidxmul);
        for (int k0 = 0; k0 < K1; k0 += 32) {
            v16h a  = load_a_f16(Arow, k0, khalf);
            v16h b0 = *(const v16h*)(b1row0 + k0 + khalf * 16);
            v16h b1 = *(const v16h*)(b1row1 + k0 + khalf * 16);
            if (k0 + 32 < K1) {
                __builtin_prefetch(b1row0 + k0 + 32, 0, 3);
                __builtin_prefetch(b1row1 + k0 + 32, 0, 3);
            }
            acc0.v = __builtin_amdgcn_wmma_f32_16x16x32_f16(
                         false, a, false, b0, (short)0, acc0.v, false, false);
            acc1.v = __builtin_amdgcn_wmma_f32_16x16x32_f16(
                         false, a, false, b1, (short)0, acc1.v, false, false);
        }
    }

    // ---- pass 2: recurrent projection (K2==0 at t==0: h_{-1}==0) ----
    {
        const _Float16* Arow   = A2 + (long long)m_global * lda2;
        const _Float16* b2row0 = B2 + (long long)n0 * ldb2;
        const _Float16* b2row1 = B2 + (long long)n1 * ldb2;
        for (int k0 = 0; k0 < K2; k0 += 32) {
            v16h a  = load_a_f16(Arow, k0, khalf);
            v16h b0 = *(const v16h*)(b2row0 + k0 + khalf * 16);
            v16h b1 = *(const v16h*)(b2row1 + k0 + khalf * 16);
            acc0.v = __builtin_amdgcn_wmma_f32_16x16x32_f16(
                         false, a, false, b0, (short)0, acc0.v, false, false);
            acc1.v = __builtin_amdgcn_wmma_f32_16x16x32_f16(
                         false, a, false, b1, (short)0, acc1.v, false, false);
        }
    }

    // ---- store both 16x16 tiles (C/D layout: row r + khalf*8, col l15) ----
    #pragma unroll
    for (int r = 0; r < 8; ++r) {
        long long crow = (long long)(mtile * 16 + khalf * 8 + r) * ldc;
        C[crow + n0] = acc0.e[r];
        C[crow + n1] = acc1.e[r];
    }
}

// ---------------------------------------------------------------------------
// f32 -> f16 weight conversion with K padding
// ---------------------------------------------------------------------------
__global__ void conv_w16(const float* __restrict__ in, _Float16* __restrict__ out,
                         int N, int K, int Kp)
{
    long long i = (long long)blockIdx.x * blockDim.x + threadIdx.x;
    if (i >= (long long)N * Kp) return;
    int n = (int)(i / Kp), k = (int)(i % Kp);
    out[i] = (k < K) ? (_Float16)in[(long long)n * K + k] : (_Float16)0.0f;
}

// ---------------------------------------------------------------------------
// Char LSTM cell: gates[32768,512] (i,f,g,o each 128 wide)
// ---------------------------------------------------------------------------
__global__ void char_cell(const float* __restrict__ gates,
                          float* __restrict__ cst, _Float16* __restrict__ h16,
                          const int* __restrict__ cl,
                          float* __restrict__ feat_out,
                          int t, int dir)
{
    int i = blockIdx.x * blockDim.x + threadIdx.x;
    if (i >= 32768 * 128) return;
    int s = i >> 7, j = i & 127;
    const float* g = gates + (long long)s * 512;
    float gi = g[j], gf = g[128 + j], gg = g[256 + j], go = g[384 + j];
    float cp = (t == 0) ? 0.0f : cst[i];
    float c  = sigmoidf_(gf) * cp + sigmoidf_(gi) * tanhf(gg);
    float h  = sigmoidf_(go) * tanhf(c);
    cst[i] = c;
    h16[i] = (_Float16)h;
    if (dir == 0) { if (t == cl[s] - 1) feat_out[i] = h; }   // last real char
    else          { if (t == 15)        feat_out[i] = h; }   // last padded step
}

// ---------------------------------------------------------------------------
// Word LSTM cell: gates[64,2048] (i,f,g,o each 512 wide). Writes masked h into
// hcat[b, pos, dir*512 + j]; pos is reversed-within-length for dir==1.
// ---------------------------------------------------------------------------
__global__ void word_cell(const float* __restrict__ gates,
                          float* __restrict__ cst, _Float16* __restrict__ h16,
                          float* __restrict__ hcat, const int* __restrict__ lens,
                          int t, int dir)
{
    int i = blockIdx.x * blockDim.x + threadIdx.x;
    if (i >= 64 * 512) return;
    int b = i >> 9, j = i & 511;
    const float* g = gates + (long long)b * 2048;
    float gi = g[j], gf = g[512 + j], gg = g[1024 + j], go = g[1536 + j];
    float cp = (t == 0) ? 0.0f : cst[i];
    float c  = sigmoidf_(gf) * cp + sigmoidf_(gi) * tanhf(gg);
    float h  = sigmoidf_(go) * tanhf(c);
    cst[i] = c;
    h16[i] = (_Float16)h;                       // recurrence carries unmasked h
    int L = lens[b];
    int pos = (dir == 0) ? t : ((t < L) ? (L - 1 - t) : t);
    float m = (pos < L) ? 1.0f : 0.0f;
    hcat[((long long)b * 512 + pos) * 1024 + dir * 512 + j] = h * m;
}

// ---------------------------------------------------------------------------
// Build word-LSTM input x16 [B,S,576]: [word_emb(300) | f(128)*m | b(128)*m | pad]
// ---------------------------------------------------------------------------
__global__ void build_x16(const int* __restrict__ sent, const int* __restrict__ lens,
                          const float* __restrict__ wemb,
                          const float* __restrict__ f_out, const float* __restrict__ b_out,
                          _Float16* __restrict__ x16)
{
    long long i = (long long)blockIdx.x * blockDim.x + threadIdx.x;
    if (i >= (long long)64 * 512 * 576) return;
    int k  = (int)(i % 576);
    int bt = (int)(i / 576);
    int b  = bt >> 9, t = bt & 511;
    float v = 0.0f;
    if (k < 300) {
        v = wemb[(long long)sent[bt] * 300 + k];
    } else if (k < 556) {
        float m = (t < lens[b]) ? 1.0f : 0.0f;
        v = (k < 428) ? f_out[(long long)bt * 128 + (k - 300)] * m
                      : b_out[(long long)bt * 128 + (k - 428)] * m;
    }
    x16[i] = (_Float16)v;
}

// Row-gather indices for the reversed-within-length backward pass: idx[t*64+b]
__global__ void build_idx(const int* __restrict__ lens, int* __restrict__ idxmat)
{
    int i = blockIdx.x * blockDim.x + threadIdx.x;
    if (i >= 512 * 64) return;
    int t = i >> 6, b = i & 63;
    int L = lens[b];
    int r = (t < L) ? (L - 1 - t) : t;
    idxmat[i] = b * 512 + r;
}

// ---------------------------------------------------------------------------
// Sequential decode with prev-label feedback: one persistent block, 640 threads
// ---------------------------------------------------------------------------
__global__ void __launch_bounds__(640)
decode_scan(const float* __restrict__ hcat, const float* __restrict__ W,
            const float* __restrict__ bias,
            float* __restrict__ out_logits, float* __restrict__ out_probs)
{
    __shared__ float prev[64][10];
    __shared__ float lg[64][10];
    int tid = threadIdx.x;
    int b = tid / 10, o = tid % 10;
    if (o < 9) prev[b][o] = 0.0f;
    __syncthreads();
    const float* wrow = W + o * 1033;
    for (int t = 0; t < 512; ++t) {
        const float* h = hcat + ((long long)b * 512 + t) * 1024;
        float acc = bias[o];
        for (int k = 0; k < 1024; k += 4)
            acc += h[k] * wrow[k] + h[k+1] * wrow[k+1]
                 + h[k+2] * wrow[k+2] + h[k+3] * wrow[k+3];
        #pragma unroll
        for (int j = 0; j < 9; ++j) acc += prev[b][j] * wrow[1024 + j];
        lg[b][o] = acc;
        out_logits[((long long)b * 512 + t) * 10 + o] = acc;
        __syncthreads();
        if (o == 0) {
            float mx = lg[b][1];
            #pragma unroll
            for (int j = 2; j < 10; ++j) mx = fmaxf(mx, lg[b][j]);
            float ex[9], s = 0.0f;
            #pragma unroll
            for (int j = 0; j < 9; ++j) { ex[j] = __expf(lg[b][1 + j] - mx); s += ex[j]; }
            float inv = 1.0f / s;
            #pragma unroll
            for (int j = 0; j < 9; ++j) {
                float pr = ex[j] * inv;
                prev[b][j] = pr;
                out_probs[((long long)b * 512 + t) * 9 + j] = pr;
            }
        }
        __syncthreads();
    }
}

// ---------------------------------------------------------------------------
extern "C" void kernel_launch(void* const* d_in, const int* in_sizes, int n_in,
                              void* d_out, int out_size, void* d_ws, size_t ws_size,
                              hipStream_t stream)
{
    (void)in_sizes; (void)n_in; (void)out_size; (void)ws_size;
    const int B = 64, S = 512, LC = 16, CE = 64, CH = 128, H = 512;
    const int Dp = 576;                 // 556 padded to multiple of 32
    const int NC = 4 * CH;              // 512
    const int NW = 4 * H;               // 2048

    const int*   sent    = (const int*)d_in[0];
    const int*   chars   = (const int*)d_in[1];
    const int*   xlens   = (const int*)d_in[2];
    const int*   clens   = (const int*)d_in[3];
    const float* wemb    = (const float*)d_in[4];
    const float* cemb    = (const float*)d_in[5];
    const float* cf_Wih  = (const float*)d_in[6];
    const float* cf_Whh  = (const float*)d_in[7];
    const float* cf_b    = (const float*)d_in[8];
    const float* cb_Wih  = (const float*)d_in[9];
    const float* cb_Whh  = (const float*)d_in[10];
    const float* cb_b    = (const float*)d_in[11];
    const float* wf_Wih  = (const float*)d_in[12];
    const float* wf_Whh  = (const float*)d_in[13];
    const float* wf_b    = (const float*)d_in[14];
    const float* wb_Wih  = (const float*)d_in[15];
    const float* wb_Whh  = (const float*)d_in[16];
    const float* wb_b    = (const float*)d_in[17];
    const float* fc_W    = (const float*)d_in[18];
    const float* fc_b    = (const float*)d_in[19];

    char* p = (char*)d_ws;
    auto carve = [&](size_t bytes) -> void* {
        void* r = (void*)p;
        p += (bytes + 255) & ~(size_t)255;
        return r;
    };
    _Float16* cfWih16 = (_Float16*)carve((size_t)NC * CE * 2);
    _Float16* cfWhh16 = (_Float16*)carve((size_t)NC * CH * 2);
    _Float16* cbWih16 = (_Float16*)carve((size_t)NC * CE * 2);
    _Float16* cbWhh16 = (_Float16*)carve((size_t)NC * CH * 2);
    _Float16* wfWih16 = (_Float16*)carve((size_t)NW * Dp * 2);
    _Float16* wfWhh16 = (_Float16*)carve((size_t)NW * H  * 2);
    _Float16* wbWih16 = (_Float16*)carve((size_t)NW * Dp * 2);
    _Float16* wbWhh16 = (_Float16*)carve((size_t)NW * H  * 2);
    float*    gates   = (float*)   carve((size_t)B * S * NC * 4);   // 67 MB, reused
    _Float16* ch_h16  = (_Float16*)carve((size_t)B * S * CH * 2);
    float*    ch_c    = (float*)   carve((size_t)B * S * CH * 4);
    float*    f_out   = (float*)   carve((size_t)B * S * CH * 4);
    float*    b_out   = (float*)   carve((size_t)B * S * CH * 4);
    _Float16* x16     = (_Float16*)carve((size_t)B * S * Dp * 2);
    int*      idxmat  = (int*)     carve((size_t)S * B * 4);
    _Float16* w_h16   = (_Float16*)carve((size_t)B * H * 2);
    float*    w_c     = (float*)   carve((size_t)B * H * 4);
    float*    hcat    = (float*)   carve((size_t)B * S * 2 * H * 4); // 134 MB

    float* out_logits = (float*)d_out;
    float* out_probs  = out_logits + (size_t)B * S * 10;

    // ---- weight conversion (f32 -> f16, K padded) ----
    auto cvt = [&](const float* in, _Float16* out, int N, int K, int Kp) {
        long long n = (long long)N * Kp;
        conv_w16<<<(unsigned)((n + 255) / 256), 256, 0, stream>>>(in, out, N, K, Kp);
    };
    cvt(cf_Wih, cfWih16, NC, CE, CE);
    cvt(cf_Whh, cfWhh16, NC, CH, CH);
    cvt(cb_Wih, cbWih16, NC, CE, CE);
    cvt(cb_Whh, cbWhh16, NC, CH, CH);
    cvt(wf_Wih, wfWih16, NW, 556, Dp);
    cvt(wf_Whh, wfWhh16, NW, H,   H);
    cvt(wb_Wih, wbWih16, NW, 556, Dp);
    cvt(wb_Whh, wbWhh16, NW, H,   H);
    build_idx<<<(512 * 64 + 255) / 256, 256, 0, stream>>>(xlens, idxmat);

    const dim3 gblk(32, 4);

    // ---- char LSTMs: M=32768 (2048 M-tiles), N=512 (16 dual-tile columns) ----
    for (int dir = 0; dir < 2; ++dir) {
        const _Float16* Wih16 = dir ? cbWih16 : cfWih16;
        const _Float16* Whh16 = dir ? cbWhh16 : cfWhh16;
        const float*    bvec  = dir ? cb_b    : cf_b;
        float*          feat  = dir ? b_out   : f_out;
        dim3 grid(NC / 32, 2048 / 4);
        for (int t = 0; t < LC; ++t) {
            // gates = bias + emb(char_t)@Wih^T + h_{t-1}@Whh^T  (one fused kernel)
            fused_wmma_gemm<2><<<grid, gblk, 0, stream>>>(
                nullptr, cemb, chars, 0, CE, 0, LC, t,
                Wih16, CE, CE,
                ch_h16, CH, Whh16, CH, (t > 0) ? CH : 0,
                gates, NC, bvec, 2048);
            char_cell<<<(32768 * 128) / 256, 256, 0, stream>>>(
                gates, ch_c, ch_h16, clens, feat, t, dir);
        }
    }

    // ---- assemble word-LSTM input ----
    build_x16<<<(unsigned)(((long long)B * S * Dp + 255) / 256), 256, 0, stream>>>(
        sent, xlens, wemb, f_out, b_out, x16);

    // ---- word BiLSTM: M=64 (4 M-tiles), N=2048 (64 dual-tile columns) ----
    for (int dir = 0; dir < 2; ++dir) {
        const _Float16* Wih16 = dir ? wbWih16 : wfWih16;
        const _Float16* Whh16 = dir ? wbWhh16 : wfWhh16;
        const float*    bvec  = dir ? wb_b    : wf_b;
        dim3 grid(NW / 32, 1);
        for (int t = 0; t < S; ++t) {
            int K2 = (t > 0) ? H : 0;
            if (dir == 0)   // rows: x16[b, t, :], stride S*Dp
                fused_wmma_gemm<0><<<grid, gblk, 0, stream>>>(
                    x16, nullptr, nullptr, (long long)t * Dp, S * Dp, 0, 0, 0,
                    Wih16, Dp, Dp,
                    w_h16, H, Whh16, H, K2,
                    gates, NW, bvec, 4);
            else            // rows gathered via reversed-within-length index
                fused_wmma_gemm<1><<<grid, gblk, 0, stream>>>(
                    x16, nullptr, idxmat + t * B, 0, 0, Dp, 0, 0,
                    Wih16, Dp, Dp,
                    w_h16, H, Whh16, H, K2,
                    gates, NW, bvec, 4);
            word_cell<<<(64 * 512) / 256, 256, 0, stream>>>(
                gates, w_c, w_h16, hcat, xlens, t, dir);
        }
    }

    // ---- sequential decode ----
    decode_scan<<<1, 640, 0, stream>>>(hcat, fc_W, fc_b, out_logits, out_probs);
}
